// MultiHeadAttention_30313879175260
// MI455X (gfx1250) — compile-verified, matrix operand order checked by reading, not run on hardware
//
#include <hip/hip_runtime.h>

// ---------------------------------------------------------------------------
// MultiHeadAttention forward for MI455X (gfx1250), wave32 + WMMA bf16.
//  B=4, S=2048, E=1024, H=16, d=64.  f32 in/out, bf16 WMMA internally.
//
// Pipeline:
//   1) wt_kernel   : W[k][n] -> WT[n][k] bf16 (so GEMM B frags are contiguous)
//   2) proj_kernel : Q/K/V projections, bf16 out.  Q,K stored [B,H,S,d];
//                    V stored TRANSPOSED [B,H,d,S] so the PV GEMM's B
//                    fragments are contiguous global b128 loads.
//   3) attn_kernel : flash attention, online softmax, O accumulated f32.
// ---------------------------------------------------------------------------

#define EMB   1024
#define HEADS 16
#define HDIM  64
#define BATCH 4
#define SEQ   2048

typedef __attribute__((ext_vector_type(16))) __bf16         v16bf;
typedef __attribute__((ext_vector_type(16))) unsigned short v16us;
typedef __attribute__((ext_vector_type(8)))  float          v8f;

union Frag {
    v16us us;
    v16bf bf;
    uint4 q[2];
};

// native f32 -> bf16 (v_cvt_pk_bf16_f32), bits out
__device__ __forceinline__ unsigned short bf_cvt(float f) {
    return __builtin_bit_cast(unsigned short, (__bf16)f);
}

// load a 16-element bf16 fragment as two b128 from two 16B-aligned runs
__device__ __forceinline__ v16bf ld_frag(const unsigned short* r0,
                                         const unsigned short* r1) {
    Frag f;
    f.q[0] = *(const uint4*)r0;
    f.q[1] = *(const uint4*)r1;
    return f.bf;
}

__device__ __forceinline__ v8f wmma_bf16(v16bf a, v16bf b, v8f c) {
    return __builtin_amdgcn_wmma_f32_16x16x32_bf16(
        /*neg_a=*/false, a, /*neg_b=*/false, b,
        /*c_mod=*/(short)0, c, /*reuse_a=*/false, /*reuse_b=*/false);
}

// ---------------------------------------------------------------------------
// Workspace layout (units: unsigned short / bf16 elements)
//   [0 .. 3*2^20)       : WqT, WkT, WvT  (each EMB*EMB, [n][k])
//   [3*2^20 .. +3*2^23) : QP [B,H,S,d], KP [B,H,S,d], VT [B,H,d,S]
// ---------------------------------------------------------------------------
#define WT_ELEMS   ((size_t)EMB * EMB)                   // 1<<20
#define PROJ_ELEMS ((size_t)BATCH * HEADS * SEQ * HDIM)  // 1<<23
#define PROJ_BASE  (3 * WT_ELEMS)

// ---------------------------------------------------------------------------
// Kernel 1: transpose + convert weights  WT[n][k] = bf16(W[k][n])
// ---------------------------------------------------------------------------
__global__ __launch_bounds__(256) void wt_kernel(const float* __restrict__ Wq,
                                                 const float* __restrict__ Wk,
                                                 const float* __restrict__ Wv,
                                                 unsigned short* __restrict__ ws) {
    const float* W = (blockIdx.z == 0) ? Wq : (blockIdx.z == 1 ? Wk : Wv);
    unsigned short* WT = ws + (size_t)blockIdx.z * WT_ELEMS;

    __shared__ float tile[32][33];
    const int x0 = blockIdx.x * 32;   // n
    const int y0 = blockIdx.y * 32;   // k
    for (int i = threadIdx.y; i < 32; i += 8)
        tile[i][threadIdx.x] = W[(size_t)(y0 + i) * EMB + x0 + threadIdx.x];
    __syncthreads();
    for (int i = threadIdx.y; i < 32; i += 8)
        WT[(size_t)(x0 + i) * EMB + y0 + threadIdx.x] =
            bf_cvt(tile[threadIdx.x][i]);
}

// ---------------------------------------------------------------------------
// Kernel 2: fused QKV projection GEMM.
//   blockIdx.z: 0=Q (src=q), 1=K (src=v), 2=V (src=v, stored transposed).
//   Each wave computes a 16(M) x 128(N) tile; K-loop step 32; 8 WMMA/step.
// ---------------------------------------------------------------------------
__global__ __launch_bounds__(256) void proj_kernel(const float* __restrict__ q,
                                                   const float* __restrict__ v,
                                                   const float* __restrict__ bq,
                                                   const float* __restrict__ bk,
                                                   const float* __restrict__ bv,
                                                   unsigned short* __restrict__ ws) {
    const int lane = threadIdx.x & 31;
    const int w    = threadIdx.x >> 5;
    const int hi   = lane >> 4;        // lane group
    const int col  = lane & 15;
    const int koff = 8 * hi;           // fragment K-group offset (ISA layout)

    const int which = blockIdx.z;
    const float* src  = (which == 0) ? q : v;
    const float* bias = (which == 0) ? bq : (which == 1 ? bk : bv);
    const unsigned short* WT = ws + (size_t)which * WT_ELEMS;
    unsigned short* dst = ws + PROJ_BASE + (size_t)which * PROJ_ELEMS;
    // layout strides: Q,K -> [.., S, d]  (s*HDIM + dc);  V -> [.., d, S]
    const int ss = (which < 2) ? HDIM : 1;
    const int cs = (which < 2) ? 1 : SEQ;

    const int m0 = blockIdx.x * 16;
    const int n0 = w * 128;

    const float* arow = src + (size_t)(m0 + col) * EMB;  // A: lane holds row m0+col

    const v8f zero = {0.f, 0.f, 0.f, 0.f, 0.f, 0.f, 0.f, 0.f};
    v8f acc[8] = {zero, zero, zero, zero, zero, zero, zero, zero};

    for (int kk = 0; kk < EMB; kk += 32) {
        // A fragment: f32 -> bf16 via native packed converts.
        // Layout: lanes 0-15 hold K{0..7,16..23}, lanes 16-31 K{8..15,24..31}.
        const float4 fa = *(const float4*)(arow + kk + koff);
        const float4 fb = *(const float4*)(arow + kk + koff + 4);
        const float4 fc = *(const float4*)(arow + kk + 16 + koff);
        const float4 fd = *(const float4*)(arow + kk + 16 + koff + 4);
        v16bf afrag;
        afrag[0]  = (__bf16)fa.x; afrag[1]  = (__bf16)fa.y;
        afrag[2]  = (__bf16)fa.z; afrag[3]  = (__bf16)fa.w;
        afrag[4]  = (__bf16)fb.x; afrag[5]  = (__bf16)fb.y;
        afrag[6]  = (__bf16)fb.z; afrag[7]  = (__bf16)fb.w;
        afrag[8]  = (__bf16)fc.x; afrag[9]  = (__bf16)fc.y;
        afrag[10] = (__bf16)fc.z; afrag[11] = (__bf16)fc.w;
        afrag[12] = (__bf16)fd.x; afrag[13] = (__bf16)fd.y;
        afrag[14] = (__bf16)fd.z; afrag[15] = (__bf16)fd.w;

#pragma unroll
        for (int nt = 0; nt < 8; ++nt) {
            // B fragment: lane holds column n; contiguous in k from WT[n][k]
            const unsigned short* wr = WT + (size_t)(n0 + nt * 16 + col) * EMB + kk;
            acc[nt] = wmma_bf16(afrag, ld_frag(wr + koff, wr + 16 + koff), acc[nt]);
        }
    }

    // Epilogue: bias add, convert, scatter.  All index components that are
    // uniform per tile are hoisted; single store path for both layouts.
    const int bi     = m0 >> 11;            // batch (tiles never straddle)
    const int s_base = (m0 & (SEQ - 1)) + 8 * hi;
#pragma unroll
    for (int nt = 0; nt < 8; ++nt) {
        const int n   = n0 + nt * 16 + col;
        const float bb = bias[n];
        const int hh  = (n0 + nt * 16) >> 6;        // head, uniform per nt
        const int dc  = ((n0 + nt * 16) & 63) + col; // channel within head
        unsigned short* drow =
            dst + (size_t)(bi * HEADS + hh) * (SEQ * HDIM) + (size_t)dc * cs;
#pragma unroll
        for (int r = 0; r < 8; ++r) {
            drow[(size_t)(s_base + r) * ss] = bf_cvt(acc[nt][r] + bb);
        }
    }
}

// ---------------------------------------------------------------------------
// Kernel 3: flash attention. One wave per (b, h, 16-row q-tile).
//   Streams keys in chunks of 32; online softmax; O accumulated in f32.
//   K and V^T fragments are contiguous global b128 loads; only the P tile
//   (C-layout -> A-layout) bounces through LDS.
// ---------------------------------------------------------------------------
__global__ __launch_bounds__(256) void attn_kernel(const unsigned short* __restrict__ ws,
                                                   float* __restrict__ out) {
    const unsigned short* QP = ws + PROJ_BASE;
    const unsigned short* KP = QP + PROJ_ELEMS;
    const unsigned short* VT = KP + PROJ_ELEMS;   // [B,H,d,S]

    __shared__ __align__(16) unsigned short Plds[8 * 16 * 32];  // 8 KB

    const int lane = threadIdx.x & 31;
    const int w    = threadIdx.x >> 5;
    const int hi   = lane >> 4;
    const int col  = lane & 15;
    const int koff = 8 * hi;

    const int wid = blockIdx.x * 8 + w;
    const int qt  = wid & 127;
    const int h   = (wid >> 7) & 15;
    const int b   = wid >> 11;

    const size_t head = (((size_t)b * HEADS + h) * SEQ) * HDIM;  // base for Q/K/VT

    // Q fragments, held in registers for the whole key loop
    const unsigned short* qrow = QP + head + (size_t)(qt * 16 + col) * HDIM;
    v16bf qa[2];
#pragma unroll
    for (int ks = 0; ks < 2; ++ks)
        qa[ks] = ld_frag(qrow + ks * 32 + koff, qrow + ks * 32 + 16 + koff);

    const v8f zero = {0.f, 0.f, 0.f, 0.f, 0.f, 0.f, 0.f, 0.f};
    v8f o[4] = {zero, zero, zero, zero};
    float mrow[8], lrow[8];
#pragma unroll
    for (int r = 0; r < 8; ++r) { mrow[r] = -3.0e38f; lrow[r] = 0.f; }

    unsigned short* pl = Plds + w * (16 * 32);
    const unsigned short* kcol = KP + head + (size_t)col * HDIM;  // key rows, this lane
    const unsigned short* vcol = VT + head + (size_t)col * SEQ;   // feature rows

    const float scale = 0.03125f;  // 1/sqrt(EMB) = 1/32, faithful to reference

    for (int key0 = 0; key0 < SEQ; key0 += 32) {
        // prefetch next chunk's K rows and V^T columns into cache
        if (key0 + 32 < SEQ) {
            __builtin_prefetch(kcol + (size_t)(key0 + 32) * HDIM, 0, 3);
            __builtin_prefetch(vcol + key0 + 32, 0, 3);
        }

        // ---- scores S = Q · K^T (16 x 32): B frags contiguous from KP rows -
        v8f s0 = zero, s1 = zero;
#pragma unroll
        for (int ks = 0; ks < 2; ++ks) {
#pragma unroll
            for (int nt = 0; nt < 2; ++nt) {
                const unsigned short* kr =
                    kcol + (size_t)(key0 + nt * 16) * HDIM + ks * 32;
                v16bf kb = ld_frag(kr + koff, kr + 16 + koff);
                if (nt == 0) s0 = wmma_bf16(qa[ks], kb, s0);
                else         s1 = wmma_bf16(qa[ks], kb, s1);
            }
        }

        // ---- online softmax (row = VGPR slot r within a 16-lane group) ----
#pragma unroll
        for (int r = 0; r < 8; ++r) {
            float sc0 = s0[r] * scale;
            float sc1 = s1[r] * scale;
            float rm = fmaxf(sc0, sc1);
            rm = fmaxf(rm, __shfl_xor(rm, 1));
            rm = fmaxf(rm, __shfl_xor(rm, 2));
            rm = fmaxf(rm, __shfl_xor(rm, 4));
            rm = fmaxf(rm, __shfl_xor(rm, 8));
            float mnew  = fmaxf(mrow[r], rm);
            float alpha = __expf(mrow[r] - mnew);
            float p0 = __expf(sc0 - mnew);
            float p1 = __expf(sc1 - mnew);
            float rs = p0 + p1;
            rs += __shfl_xor(rs, 1);
            rs += __shfl_xor(rs, 2);
            rs += __shfl_xor(rs, 4);
            rs += __shfl_xor(rs, 8);
            lrow[r] = lrow[r] * alpha + rs;
            mrow[r] = mnew;
            o[0][r] *= alpha; o[1][r] *= alpha;
            o[2][r] *= alpha; o[3][r] *= alpha;
            // C-layout -> LDS P tile [16][32] as bf16 (native cvt)
            pl[(r + 8 * hi) * 32 + col]      = bf_cvt(p0);
            pl[(r + 8 * hi) * 32 + 16 + col] = bf_cvt(p1);
        }
        __syncthreads();   // order the cross-lane P transpose

        // ---- re-load P in A-fragment layout (2 x ds_load_b128) ------------
        v16bf pa = ld_frag(pl + col * 32 + koff, pl + col * 32 + 16 + koff);

        // ---- O += P · V : B frags contiguous from V^T rows (global b128) --
#pragma unroll
        for (int nt = 0; nt < 4; ++nt) {
            const unsigned short* vr = vcol + (size_t)(nt * 16) * SEQ + key0;
            v16bf vb = ld_frag(vr + koff, vr + 16 + koff);
            o[nt] = wmma_bf16(pa, vb, o[nt]);
        }
    }

    // ---- epilogue: O *= rcp(l), store f32 to [B, S, E] --------------------
    float rinv[8];
#pragma unroll
    for (int r = 0; r < 8; ++r) rinv[r] = __builtin_amdgcn_rcpf(lrow[r]);

    float* orow = out + ((size_t)b * SEQ + qt * 16 + 8 * hi) * EMB
                      + h * HDIM + col;
#pragma unroll
    for (int nt = 0; nt < 4; ++nt) {
#pragma unroll
        for (int r = 0; r < 8; ++r) {
            orow[(size_t)r * EMB + nt * 16] = o[nt][r] * rinv[r];
        }
    }
}

// ---------------------------------------------------------------------------
extern "C" void kernel_launch(void* const* d_in, const int* in_sizes, int n_in,
                              void* d_out, int out_size, void* d_ws, size_t ws_size,
                              hipStream_t stream) {
    const float* q  = (const float*)d_in[0];
    const float* v  = (const float*)d_in[1];
    const float* Wq = (const float*)d_in[2];
    const float* bq = (const float*)d_in[3];
    const float* Wk = (const float*)d_in[4];
    const float* bk = (const float*)d_in[5];
    const float* Wv = (const float*)d_in[6];
    const float* bv = (const float*)d_in[7];
    unsigned short* ws = (unsigned short*)d_ws;
    float* out = (float*)d_out;

    // 1) weight transpose + bf16 convert
    wt_kernel<<<dim3(32, 32, 3), dim3(32, 8, 1), 0, stream>>>(Wq, Wk, Wv, ws);

    // 2) fused Q/K/V projections (M=8192, N=1024, K=1024 each)
    proj_kernel<<<dim3(512, 1, 3), 256, 0, stream>>>(q, v, bq, bk, bv, ws);

    // 3) flash attention: 8192 wave-tiles, 8 waves/block
    attn_kernel<<<dim3(1024, 1, 1), 256, 0, stream>>>(ws, out);
}
